// AttentionBlock_10385230922046
// MI455X (gfx1250) — matrix operand order repaired and checked by reading, not compiled, not run on hardware
//
#include <hip/hip_runtime.h>
#include <hip/hip_bf16.h>
#include <math.h>

// ---------------------------------------------------------------------------
// Transformer encoder block on MI455X (gfx1250, wave32, WMMA).
// All GEMMs + attention run through v_wmma_f32_16x16x32_bf16, with all WMMA
// operands fetched via contiguous 128-bit loads (weights pre-transposed,
// V stored [B,H,DH,S]).  Requires ~232 MB of workspace in d_ws.
// ---------------------------------------------------------------------------

typedef __attribute__((ext_vector_type(16))) __bf16 v16bf;
typedef __attribute__((ext_vector_type(8)))  __bf16 bf16x8;
typedef __attribute__((ext_vector_type(8)))  float  v8f;

static constexpr int B_  = 8;
static constexpr int S_  = 1024;
static constexpr int D_  = 1024;
static constexpr int H_  = 16;
static constexpr int F_  = 4096;
static constexpr int DH_ = 64;
#define LN_EPS 1e-3f

__device__ __forceinline__ int lane_id() { return threadIdx.x & 31; }

// ---- WMMA operand loaders --------------------------------------------------
// A-operand 16x32 bf16 from a ROW-MAJOR 16x32 tile (contiguous k order:
// lane (r,h) element e -> K = 16h+e). Per lane this is one
// contiguous 16-element (32B) run of row r -> two b128 loads.
__device__ __forceinline__ v16bf load_A_rowmajor(const __bf16* base, int ld) {
  const int l = lane_id(), r = l & 15, h = l >> 4;
  const __bf16* p = base + (size_t)r * ld + 16 * h;
  bf16x8 lo = *(const bf16x8*)(p);
  bf16x8 hi = *(const bf16x8*)(p + 8);
  return __builtin_shufflevector(lo, hi, 0, 1, 2, 3, 4, 5, 6, 7,
                                 8, 9, 10, 11, 12, 13, 14, 15);
}

// B-operand 32x16 (K x N) whose SOURCE is row-major [N x K] (i.e. transposed):
// lane (c,h) element e -> K=16h+e, N=c  ==>  16 contiguous elements (32B).
__device__ __forceinline__ v16bf load_B_from_T(const __bf16* baseT, int ld) {
  const int l = lane_id(), c = l & 15, h = l >> 4;
  return *(const v16bf*)(baseT + (size_t)c * ld + 16 * h);
}

__device__ __forceinline__ v8f wmma_bf16(v16bf a, v16bf b, v8f c) {
  return __builtin_amdgcn_wmma_f32_16x16x32_bf16(false, a, false, b, (short)0, c,
                                                 false, false);
}

// ---- fp32 [K x N] -> bf16 [N x K] transposed conversion --------------------
__global__ void cvt_transpose_kernel(const float* __restrict__ src,
                                     __bf16* __restrict__ dst, int Kd, int Nd) {
  int i = blockIdx.x * blockDim.x + threadIdx.x;
  int stride = gridDim.x * blockDim.x;
  int total = Kd * Nd;
  for (; i < total; i += stride) {
    int n = i / Kd, kk = i - n * Kd;
    dst[i] = (__bf16)src[(size_t)kk * Nd + n];
  }
}

// ---- block reduction helper ------------------------------------------------
__device__ __forceinline__ float block_sum(float val, float* sbuf) {
  int l = threadIdx.x & 31, wid = threadIdx.x >> 5;
#pragma unroll
  for (int o = 16; o > 0; o >>= 1) val += __shfl_xor(val, o);
  if (l == 0) sbuf[wid] = val;
  __syncthreads();
  float r = 0.f;
  if (threadIdx.x < (blockDim.x >> 5)) r = sbuf[threadIdx.x];
  if (wid == 0) {
#pragma unroll
    for (int o = 16; o > 0; o >>= 1) r += __shfl_xor(r, o);
    if (l == 0) sbuf[0] = r;
  }
  __syncthreads();
  r = sbuf[0];
  __syncthreads();
  return r;
}

// ---- LayerNorm: fp32 row -> bf16 row --------------------------------------
__global__ __launch_bounds__(256) void ln_f32_to_bf16_kernel(
    const float* __restrict__ x, const float* __restrict__ g,
    const float* __restrict__ bb, __bf16* __restrict__ out, int n) {
  __shared__ float sbuf[32];
  const size_t rb = (size_t)blockIdx.x * n;
  float s = 0.f, s2 = 0.f;
  for (int i = threadIdx.x; i < n; i += blockDim.x) {
    float v = x[rb + i];
    s += v; s2 += v * v;
  }
  s  = block_sum(s, sbuf);
  s2 = block_sum(s2, sbuf);
  float m = s / n;
  float rstd = rsqrtf(s2 / n - m * m + LN_EPS);
  for (int i = threadIdx.x; i < n; i += blockDim.x) {
    out[rb + i] = (__bf16)((x[rb + i] - m) * rstd * g[i] + bb[i]);
  }
}

// ---- LayerNorm: bf16 row -> bf16 row (post-fc norm over F) -----------------
__global__ __launch_bounds__(256) void ln_bf16_to_bf16_kernel(
    const __bf16* __restrict__ x, const float* __restrict__ g,
    const float* __restrict__ bb, __bf16* __restrict__ out, int n) {
  __shared__ float sbuf[32];
  const size_t rb = (size_t)blockIdx.x * n;
  float s = 0.f, s2 = 0.f;
  for (int i = threadIdx.x; i < n; i += blockDim.x) {
    float v = (float)x[rb + i];
    s += v; s2 += v * v;
  }
  s  = block_sum(s, sbuf);
  s2 = block_sum(s2, sbuf);
  float m = s / n;
  float rstd = rsqrtf(s2 / n - m * m + LN_EPS);
  for (int i = threadIdx.x; i < n; i += blockDim.x) {
    float v = (float)x[rb + i];
    out[rb + i] = (__bf16)((v - m) * rstd * g[i] + bb[i]);
  }
}

// ---- fused: x2 = LN(y)+x ; h2 = LN(x2)  (post-attn + pre-fc norms) ---------
__global__ __launch_bounds__(256) void ln_add_ln_kernel(
    const float* __restrict__ y, const float* __restrict__ x,
    const float* __restrict__ g1, const float* __restrict__ b1,
    const float* __restrict__ g2, const float* __restrict__ b2,
    float* __restrict__ x2, __bf16* __restrict__ h2, int n) {
  __shared__ float sbuf[32];
  const size_t rb = (size_t)blockIdx.x * n;
  float s = 0.f, s2 = 0.f;
  for (int i = threadIdx.x; i < n; i += blockDim.x) {
    float v = y[rb + i];
    s += v; s2 += v * v;
  }
  s  = block_sum(s, sbuf);
  s2 = block_sum(s2, sbuf);
  float m = s / n, rstd = rsqrtf(s2 / n - m * m + LN_EPS);
  float ts = 0.f, ts2 = 0.f;
  for (int i = threadIdx.x; i < n; i += blockDim.x) {
    float t = (y[rb + i] - m) * rstd * g1[i] + b1[i] + x[rb + i];
    x2[rb + i] = t;
    ts += t; ts2 += t * t;
  }
  ts  = block_sum(ts, sbuf);
  ts2 = block_sum(ts2, sbuf);
  float m2 = ts / n, rstd2 = rsqrtf(ts2 / n - m2 * m2 + LN_EPS);
  for (int i = threadIdx.x; i < n; i += blockDim.x) {
    float t = x2[rb + i];
    h2[rb + i] = (__bf16)((t - m2) * rstd2 * g2[i] + b2[i]);
  }
}

// ---- tiled WMMA GEMM: C[MxN] = A[MxK](bf16) @ Wt[NxK](bf16,transposed) -----
// block = 128 threads = 4 waves; block tile 64(M) x 128(N);
// each wave: 4 M-tiles x 2 N-tiles = 8 WMMA accumulators.
// mode 0: outH = bf16(acc * scale)                         (row-major [M,N])
// mode 1: outF = acc
// mode 2: outH = bf16(gelu(acc + bias))                    (exact gelu)
// mode 3: outF = acc + bias + resid * lam
// mode 4: outH = bf16(acc) scattered to V^T layout [B,H,DH,S]
__global__ __launch_bounds__(128) void gemm_bf16_kernel(
    const __bf16* __restrict__ A, const __bf16* __restrict__ Wt,
    int M, int N, int K, int mode, float scale,
    const float* __restrict__ bias, const float* __restrict__ resid,
    const float* __restrict__ lam, float* __restrict__ outF,
    __bf16* __restrict__ outH) {
  const int w = threadIdx.x >> 5;
  const int l = lane_id(), cc = l & 15, hh = l >> 4;
  const int n0 = blockIdx.x * 128 + w * 32;
  const int m0 = blockIdx.y * 64;
  v8f acc[4][2] = {};
  for (int kc = 0; kc < K; kc += 32) {
    v16bf bt0 = load_B_from_T(Wt + (size_t)n0 * K + kc, K);
    v16bf bt1 = load_B_from_T(Wt + (size_t)(n0 + 16) * K + kc, K);
    v16bf a0 = load_A_rowmajor(A + (size_t)(m0 +  0) * K + kc, K);
    v16bf a1 = load_A_rowmajor(A + (size_t)(m0 + 16) * K + kc, K);
    v16bf a2 = load_A_rowmajor(A + (size_t)(m0 + 32) * K + kc, K);
    v16bf a3 = load_A_rowmajor(A + (size_t)(m0 + 48) * K + kc, K);
    acc[0][0] = wmma_bf16(a0, bt0, acc[0][0]);
    acc[0][1] = wmma_bf16(a0, bt1, acc[0][1]);
    acc[1][0] = wmma_bf16(a1, bt0, acc[1][0]);
    acc[1][1] = wmma_bf16(a1, bt1, acc[1][1]);
    acc[2][0] = wmma_bf16(a2, bt0, acc[2][0]);
    acc[2][1] = wmma_bf16(a2, bt1, acc[2][1]);
    acc[3][0] = wmma_bf16(a3, bt0, acc[3][0]);
    acc[3][1] = wmma_bf16(a3, bt1, acc[3][1]);
  }
#pragma unroll
  for (int mt = 0; mt < 4; ++mt) {
#pragma unroll
    for (int nt = 0; nt < 2; ++nt) {
#pragma unroll
      for (int j = 0; j < 8; ++j) {
        int row = m0 + mt * 16 + j + 8 * hh;
        int col = n0 + nt * 16 + cc;
        float v = acc[mt][nt][j];
        size_t idx = (size_t)row * N + col;
        if (mode == 0) {
          outH[idx] = (__bf16)(v * scale);
        } else if (mode == 1) {
          outF[idx] = v;
        } else if (mode == 2) {
          v += bias[col];
          v = 0.5f * v * (1.0f + erff(v * 0.70710678118654752f));
          outH[idx] = (__bf16)v;
        } else if (mode == 3) {
          v += bias[col];
          v += resid[idx] * lam[col];
          outF[idx] = v;
        } else {  // mode 4: scatter into V^T [B, H, DH, S]
          int b = row >> 10, sidx = row & (S_ - 1);
          int head = col >> 6, dh = col & (DH_ - 1);
          outH[(((size_t)b * H_ + head) * DH_ + dh) * S_ + sidx] = (__bf16)v;
        }
      }
    }
  }
}

// ---- flash attention: per wave one 16-query tile, stream 32-key chunks -----
// q is pre-scaled by 1/sqrt(DH); k is [B,S,H,DH]; v is transposed [B,H,DH,S].
// o = softmax(q k^T + mask) @ v, scaled per-head by gamma.
__global__ __launch_bounds__(128) void flash_attn_kernel(
    const __bf16* __restrict__ q, const __bf16* __restrict__ k,
    const __bf16* __restrict__ vt, const int* __restrict__ kpm,
    const float* __restrict__ gamma, __bf16* __restrict__ o) {
  __shared__ __bf16 pbuf[4][16 * 32];
  const int w = threadIdx.x >> 5;
  const int l = lane_id(), cc = l & 15, hh = l >> 4;
  const int b = blockIdx.z, head = blockIdx.y;
  const int q0 = blockIdx.x * 64 + w * 16;
  const size_t baseQ = ((size_t)b * S_ + q0) * D_ + head * DH_;
  v16bf aq0 = load_A_rowmajor(q + baseQ, D_);
  v16bf aq1 = load_A_rowmajor(q + baseQ + 32, D_);
  const __bf16* vtb = vt + ((size_t)b * H_ + head) * DH_ * S_;
  v8f outAcc[4] = {};
  float mprev[8], lsum[8];
#pragma unroll
  for (int j = 0; j < 8; ++j) { mprev[j] = -3.0e38f; lsum[j] = 0.f; }

  for (int kc = 0; kc < S_; kc += 32) {
    v8f s[2];
#pragma unroll
    for (int t = 0; t < 2; ++t) {
      int kb = kc + t * 16;
      const __bf16* kptr = k + ((size_t)b * S_ + kb) * D_ + head * DH_;
      v16bf bk0 = load_B_from_T(kptr, D_);       // K rows are contiguous dh
      v16bf bk1 = load_B_from_T(kptr + 32, D_);
      v8f z = {};
      z = wmma_bf16(aq0, bk0, z);
      s[t] = wmma_bf16(aq1, bk1, z);
      // key_padding_mask: every element in this lane shares key column kb+cc
      if (kpm[b * S_ + kb + cc]) {
#pragma unroll
        for (int j = 0; j < 8; ++j) s[t][j] = -1e9f;
      }
    }
    // online softmax: C-layout row j+8h lives across the 16 lanes of one half
#pragma unroll
    for (int j = 0; j < 8; ++j) {
      float m = fmaxf(s[0][j], s[1][j]);
      m = fmaxf(m, __shfl_xor(m, 1));
      m = fmaxf(m, __shfl_xor(m, 2));
      m = fmaxf(m, __shfl_xor(m, 4));
      m = fmaxf(m, __shfl_xor(m, 8));
      float mnew = fmaxf(mprev[j], m);
      float alpha = __expf(mprev[j] - mnew);
      float p0 = __expf(s[0][j] - mnew);
      float p1 = __expf(s[1][j] - mnew);
      s[0][j] = p0; s[1][j] = p1;
      float rs = p0 + p1;
      rs += __shfl_xor(rs, 1);
      rs += __shfl_xor(rs, 2);
      rs += __shfl_xor(rs, 4);
      rs += __shfl_xor(rs, 8);
      lsum[j] = lsum[j] * alpha + rs;
      mprev[j] = mnew;
#pragma unroll
      for (int d = 0; d < 4; ++d) outAcc[d][j] *= alpha;
    }
    // stage P through LDS (row-major 16x32) to re-layout into A-operand format
#pragma unroll
    for (int t = 0; t < 2; ++t)
#pragma unroll
      for (int j = 0; j < 8; ++j)
        pbuf[w][(j + 8 * hh) * 32 + t * 16 + cc] = (__bf16)s[t][j];
    v16bf ap = load_A_rowmajor(&pbuf[w][0], 32);  // ds_load_b128 x2, in-order
#pragma unroll
    for (int d = 0; d < 4; ++d) {
      v16bf bv = load_B_from_T(vtb + (size_t)(d * 16) * S_ + kc, S_);
      outAcc[d] = wmma_bf16(ap, bv, outAcc[d]);
    }
  }
  float g = gamma[head];
#pragma unroll
  for (int j = 0; j < 8; ++j) {
    float inv = (lsum[j] > 0.f) ? (g / lsum[j]) : 0.f;
    int qrow = q0 + j + 8 * hh;
#pragma unroll
    for (int d = 0; d < 4; ++d)
      o[((size_t)b * S_ + qrow) * D_ + head * DH_ + d * 16 + cc] =
          (__bf16)(outAcc[d][j] * inv);
  }
}

// ---------------------------------------------------------------------------
extern "C" void kernel_launch(void* const* d_in, const int* in_sizes, int n_in,
                              void* d_out, int out_size, void* d_ws,
                              size_t ws_size, hipStream_t stream) {
  (void)in_sizes; (void)n_in; (void)out_size; (void)ws_size;
  const float* x    = (const float*)d_in[0];
  const int*   kpm  = (const int*)d_in[1];
  const float* wq   = (const float*)d_in[2];
  const float* wk   = (const float*)d_in[3];
  const float* wv   = (const float*)d_in[4];
  const float* wo   = (const float*)d_in[5];
  const float* gam  = (const float*)d_in[6];
  const float* ln_g = (const float*)d_in[7];
  const float* ln_b = (const float*)d_in[8];
  const float* fcg  = (const float*)d_in[9];
  const float* fcb  = (const float*)d_in[10];
  const float* w1   = (const float*)d_in[11];
  const float* b1   = (const float*)d_in[12];
  const float* w2   = (const float*)d_in[13];
  const float* b2   = (const float*)d_in[14];
  const float* lam  = (const float*)d_in[15];
  float* out = (float*)d_out;

  char* ws = (char*)d_ws;
  const size_t MB = 1024ull * 1024ull;
  __bf16* wq_t  = (__bf16*)(ws + 0 * MB);    //  2 MB  (all weights transposed)
  __bf16* wk_t  = (__bf16*)(ws + 2 * MB);    //  2 MB
  __bf16* wv_t  = (__bf16*)(ws + 4 * MB);    //  2 MB
  __bf16* wo_t  = (__bf16*)(ws + 6 * MB);    //  2 MB
  __bf16* w1_t  = (__bf16*)(ws + 8 * MB);    //  8 MB
  __bf16* w2_t  = (__bf16*)(ws + 16 * MB);   //  8 MB
  __bf16* h_bf  = (__bf16*)(ws + 24 * MB);   // 16 MB (reused for h2)
  __bf16* q_bf  = (__bf16*)(ws + 40 * MB);   // 16 MB
  __bf16* k_bf  = (__bf16*)(ws + 56 * MB);   // 16 MB
  __bf16* vt_bf = (__bf16*)(ws + 72 * MB);   // 16 MB (V^T: [B,H,DH,S])
  __bf16* a_bf  = (__bf16*)(ws + 88 * MB);   // 16 MB (gamma-scaled attn out)
  float*  y_f   = (float*)(ws + 104 * MB);   // 32 MB (attn projection, f32)
  float*  x2_f  = (float*)(ws + 136 * MB);   // 32 MB (mid residual, f32)
  __bf16* fc_bf = (__bf16*)(ws + 168 * MB);  // 64 MB (gelu(fc1) bf16)
  __bf16* h3_bf = (__bf16*)(ws + 40 * MB);   // 64 MB, reuses dead q/k/v/attn

  const int BS = B_ * S_;
  const dim3 blk256(256), blk128(128);

  // 1) weights -> bf16, transposed to [N x K] for contiguous B-operand loads
  cvt_transpose_kernel<<<(D_ * D_) / 256, blk256, 0, stream>>>(wq, wq_t, D_, D_);
  cvt_transpose_kernel<<<(D_ * D_) / 256, blk256, 0, stream>>>(wk, wk_t, D_, D_);
  cvt_transpose_kernel<<<(D_ * D_) / 256, blk256, 0, stream>>>(wv, wv_t, D_, D_);
  cvt_transpose_kernel<<<(D_ * D_) / 256, blk256, 0, stream>>>(wo, wo_t, D_, D_);
  cvt_transpose_kernel<<<(D_ * F_) / 256, blk256, 0, stream>>>(w1, w1_t, D_, F_);
  cvt_transpose_kernel<<<(F_ * D_) / 256, blk256, 0, stream>>>(w2, w2_t, F_, D_);

  // 2) pre-attn LN -> bf16
  ln_f32_to_bf16_kernel<<<BS, blk256, 0, stream>>>(x, ln_g, ln_b, h_bf, D_);

  // 3) Q/K/V projections (1/sqrt(DH)=0.125 folded into Q; V written transposed)
  dim3 gProj(D_ / 128, BS / 64);
  gemm_bf16_kernel<<<gProj, blk128, 0, stream>>>(h_bf, wq_t, BS, D_, D_, 0,
      0.125f, nullptr, nullptr, nullptr, nullptr, q_bf);
  gemm_bf16_kernel<<<gProj, blk128, 0, stream>>>(h_bf, wk_t, BS, D_, D_, 0,
      1.0f, nullptr, nullptr, nullptr, nullptr, k_bf);
  gemm_bf16_kernel<<<gProj, blk128, 0, stream>>>(h_bf, wv_t, BS, D_, D_, 4,
      1.0f, nullptr, nullptr, nullptr, nullptr, vt_bf);

  // 4) flash attention (+ per-head gamma)
  dim3 gAttn(S_ / 64, H_, B_);
  flash_attn_kernel<<<gAttn, blk128, 0, stream>>>(q_bf, k_bf, vt_bf, kpm, gam,
                                                  a_bf);

  // 5) output projection -> f32
  gemm_bf16_kernel<<<gProj, blk128, 0, stream>>>(a_bf, wo_t, BS, D_, D_, 1,
      1.0f, nullptr, nullptr, nullptr, y_f, nullptr);

  // 6) x2 = LN(y)+x ; h2 = LN(x2) -> bf16 (reuse h_bf)
  ln_add_ln_kernel<<<BS, blk256, 0, stream>>>(y_f, x, ln_g + D_, ln_b + D_,
      ln_g + 2 * D_, ln_b + 2 * D_, x2_f, h_bf, D_);

  // 7) FFN up-projection + exact GELU -> bf16
  dim3 gFc1(F_ / 128, BS / 64);
  gemm_bf16_kernel<<<gFc1, blk128, 0, stream>>>(h_bf, w1_t, BS, F_, D_, 2,
      1.0f, b1, nullptr, nullptr, nullptr, fc_bf);

  // 8) post-fc LN over F -> bf16
  ln_bf16_to_bf16_kernel<<<BS, blk256, 0, stream>>>(fc_bf, fcg, fcb, h3_bf, F_);

  // 9) FFN down-projection + bias + resid*lam -> d_out (f32)
  dim3 gFc2(D_ / 128, BS / 64);
  gemm_bf16_kernel<<<gFc2, blk128, 0, stream>>>(h3_bf, w2_t, BS, D_, F_, 3,
      1.0f, b2, x2_f, lam, out, nullptr);
}
